// HDNBlock_14061722927151
// MI455X (gfx1250) — compile-verified, matrix operand order checked by reading, not run on hardware
//
#include <hip/hip_runtime.h>

// CDNA5 / gfx1250 implementation of the HDN GAT block.
// GEMMs use v_wmma_f32_16x16x32_bf16 (wave32 WMMA), K padded 200->224, N 100->112.
// Segment softmax / scatter aggregation via float global atomics.
// Graph-layernorm stats via LDS (ds_add_f32) reduction into 512 buckets.

typedef __attribute__((ext_vector_type(16))) __bf16 v16bf;
typedef __attribute__((ext_vector_type(8)))  __bf16 v8bf;
typedef __attribute__((ext_vector_type(8)))  float  v8f;

#define DF   200   // feature dim
#define HCF  100   // H*C = 2*50
#define KP   224   // padded K (7 * 32)
#define NPAD 112   // padded N (7 * 16)
#define NGR  512   // graphs per batch

// ---------------- helpers ----------------

__device__ __forceinline__ __bf16 f2bf(float f) {
  unsigned u = __float_as_uint(f);
  unsigned r = (u + 0x7FFFu + ((u >> 16) & 1u)) >> 16;   // round-to-nearest-even
  unsigned short s = (unsigned short)r;
  __bf16 b;
  __builtin_memcpy(&b, &s, sizeof(b));
  return b;
}

__device__ __forceinline__ float eluf(float x) { return x > 0.f ? x : __expf(x) - 1.f; }

__device__ __forceinline__ void atomicMaxF(float* addr, float v) {
  // monotonic IEEE bit trick: works for mixed signs, init must be -inf
  if (v >= 0.f) atomicMax((int*)addr, __float_as_int(v));
  else          atomicMin((unsigned int*)addr, __float_as_uint(v));
}

// ---------------- prep kernels ----------------

__global__ void fill_kernel(float* __restrict__ p, float v, long n) {
  long i = (long)blockIdx.x * blockDim.x + threadIdx.x;
  if (i < n) p[i] = v;
}

// x[N,200] f32 -> xb[N,224] bf16 (zero pad), optional elu
__global__ void cast_pad_kernel(const float* __restrict__ x, __bf16* __restrict__ xb,
                                long N, int do_elu) {
  long i = (long)blockIdx.x * blockDim.x + threadIdx.x;
  if (i >= N * KP) return;
  long n = i / KP;
  int  k = (int)(i % KP);
  float v = 0.f;
  if (k < DF) { v = x[n * DF + k]; if (do_elu) v = eluf(v); }
  xb[i] = f2bf(v);
}

// W[200,100] f32 -> Wt[112,224] bf16 where Wt[n][k] = W[k][n] (zero padded)
__global__ void prep_wt_kernel(const float* __restrict__ W, __bf16* __restrict__ Wt) {
  int i = blockIdx.x * blockDim.x + threadIdx.x;
  if (i >= NPAD * KP) return;
  int n = i / KP, k = i % KP;
  float v = (n < HCF && k < DF) ? W[k * HCF + n] : 0.f;
  Wt[i] = f2bf(v);
}

// wv[d,h] = sum_c W[d, h*50+c] * a[h,c]   (fused attention vector: (x@W)·a = x@wv)
__global__ void prep_wv_kernel(const float* __restrict__ W, const float* __restrict__ a,
                               float* __restrict__ wv) {
  int i = blockIdx.x * blockDim.x + threadIdx.x;
  if (i >= DF * 2) return;
  int d = i >> 1, h = i & 1;
  float s = 0.f;
  for (int c = 0; c < 50; ++c) s += W[d * HCF + h * 50 + c] * a[h * 50 + c];
  wv[d * 2 + h] = s;
}

// al[n,h] = sum_d f(x[n,d]) * wv[d,h]
__global__ void al_kernel(const float* __restrict__ x, const float* __restrict__ wv,
                          float* __restrict__ al, int N, int do_elu) {
  int i = blockIdx.x * blockDim.x + threadIdx.x;
  if (i >= 2 * N) return;
  int n = i >> 1, h = i & 1;
  const float* xr = x + (long)n * DF;
  float s = 0.f;
  for (int d = 0; d < DF; ++d) {
    float v = xr[d];
    if (do_elu) v = eluf(v);
    s += v * wv[d * 2 + h];
  }
  al[i] = s;
}

// xcat[n, f] = f<100 ? b0[f] : b1[f-100]   (bias pre-init of concat buffer)
__global__ void init_xcat_kernel(float* __restrict__ x, const float* __restrict__ b0,
                                 const float* __restrict__ b1, long n) {
  long i = (long)blockIdx.x * blockDim.x + threadIdx.x;
  if (i >= n) return;
  int f = (int)(i % DF);
  x[i] = (f < HCF) ? b0[f] : b1[f - HCF];
}

// ---------------- WMMA GEMM: C[N,100] = A_bf16[N,224] @ Wt_bf16[112,224]^T ----------------
// 4 waves / block, each wave owns a 16-row x 112-col tile. K loop: 7 steps of 32.
// A fragment per-lane layout (ISA 7.12.2, 16-bit A 16x32):
//   element i (of 16): K = (i/8)*16 + (lane/16)*8 + (i%8),  row M = lane & 15
// B fragment == A-style load of Wt rows (column-major B).

__device__ __forceinline__ v16bf load_frag(const __bf16* __restrict__ row, int kb, int hf) {
  v8bf lo = *(const v8bf*)(row + kb + hf * 8);
  v8bf hi = *(const v8bf*)(row + kb + 16 + hf * 8);
  v16bf r;
#pragma unroll
  for (int i = 0; i < 8; ++i) { r[i] = lo[i]; r[i + 8] = hi[i]; }
  return r;
}

__global__ void gemm_bf16_kernel(const __bf16* __restrict__ A, const __bf16* __restrict__ Bt,
                                 float* __restrict__ C, int Nrows) {
  int wave = threadIdx.x >> 5;
  int lane = threadIdx.x & 31;
  int m0 = (blockIdx.x * 4 + wave) * 16;
  if (m0 >= Nrows) return;
  int hf = lane >> 4;
  int r  = lane & 15;
  int row = m0 + r; if (row >= Nrows) row = Nrows - 1;
  const __bf16* arow = A + (long)row * KP;

  v8f acc[7];
  v8f zero = {};
#pragma unroll
  for (int j = 0; j < 7; ++j) acc[j] = zero;

#pragma unroll
  for (int kt = 0; kt < 7; ++kt) {
    int kb = kt * 32;
    v16bf afrag = load_frag(arow, kb, hf);
#pragma unroll
    for (int j = 0; j < 7; ++j) {
      const __bf16* brow = Bt + (long)(j * 16 + r) * KP;
      v16bf bfrag = load_frag(brow, kb, hf);
      acc[j] = __builtin_amdgcn_wmma_f32_16x16x32_bf16(
          false, afrag, false, bfrag, (short)0, acc[j], false, false);
    }
  }

  // D layout: VGPR v -> M = m0 + v + 8*hf, N = j*16 + r
#pragma unroll
  for (int j = 0; j < 7; ++j) {
#pragma unroll
    for (int v = 0; v < 8; ++v) {
      int mm = m0 + v + 8 * hf;
      int nn = j * 16 + r;
      if (mm < Nrows && nn < HCF) C[(long)mm * HCF + nn] = acc[j][v];
    }
  }
}

// ---------------- edge softmax / aggregation ----------------

__global__ void edge_logit_max_kernel(const int* __restrict__ sidx, const int* __restrict__ didx,
                                      int E, int nl, const float* __restrict__ als,
                                      const float* __restrict__ ald,
                                      float* __restrict__ elog, float* __restrict__ m) {
  long i = (long)blockIdx.x * blockDim.x + threadIdx.x;
  long Et2 = 2L * ((long)E + nl);
  if (i >= Et2) return;
  long e = i >> 1; int h = (int)(i & 1);
  int s = (e < E) ? sidx[e] : (int)(e - E);
  int d = (e < E) ? didx[e] : (int)(e - E);
  float l = als[(long)s * 2 + h] + ald[(long)d * 2 + h];
  l = l > 0.f ? l : 0.2f * l;               // leaky_relu 0.2
  elog[i] = l;
  atomicMaxF(m + (long)d * 2 + h, l);
}

__global__ void edge_expsum_kernel(const int* __restrict__ didx, int E, int nl,
                                   float* __restrict__ elog, const float* __restrict__ m,
                                   float* __restrict__ z) {
  long i = (long)blockIdx.x * blockDim.x + threadIdx.x;
  long Et2 = 2L * ((long)E + nl);
  if (i >= Et2) return;
  long e = i >> 1; int h = (int)(i & 1);
  int d = (e < E) ? didx[e] : (int)(e - E);
  float ex = __expf(elog[i] - m[(long)d * 2 + h]);
  elog[i] = ex;
  atomicAdd(z + (long)d * 2 + h, ex);
}

// one 256-thread block handles 2 edges x 128 lanes (100 active features each)
__global__ void edge_agg_kernel(const int* __restrict__ sidx, const int* __restrict__ didx,
                                int E, int nl, const float* __restrict__ elog,
                                const float* __restrict__ z, const float* __restrict__ hs,
                                float* __restrict__ outp, int col) {
  long e = (long)blockIdx.x * 2 + (threadIdx.x >> 7);
  int f = threadIdx.x & 127;
  if (e >= (long)E + nl || f >= HCF) return;
  int s = (e < E) ? sidx[e] : (int)(e - E);
  int d = (e < E) ? didx[e] : (int)(e - E);
  int h = (f >= 50) ? 1 : 0;
  float alpha = elog[e * 2 + h] / (z[(long)d * 2 + h] + 1e-16f);
  atomicAdd(outp + (long)d * DF + col + f, hs[(long)s * HCF + f] * alpha);
}

// ---------------- graph layernorm ----------------

__global__ void ln_stats_kernel(const float* __restrict__ xcat, const int* __restrict__ batch,
                                int N, float* __restrict__ s1, float* __restrict__ s2,
                                float* __restrict__ cnt) {
  __shared__ float ls1[NGR], ls2[NGR], lc[NGR];
  for (int i = threadIdx.x; i < NGR; i += blockDim.x) { ls1[i] = 0.f; ls2[i] = 0.f; lc[i] = 0.f; }
  __syncthreads();
  int n = blockIdx.x * blockDim.x + threadIdx.x;
  if (n < N) {
    const float* xr = xcat + (long)n * DF;
    float a = 0.f, b = 0.f;
    for (int f = 0; f < DF; ++f) { float v = xr[f]; a += v; b += v * v; }
    int g = batch[n];
    atomicAdd(&ls1[g], a); atomicAdd(&ls2[g], b); atomicAdd(&lc[g], 1.f);
  }
  __syncthreads();
  for (int i = threadIdx.x; i < NGR; i += blockDim.x) {
    if (lc[i] != 0.f) { atomicAdd(&s1[i], ls1[i]); atomicAdd(&s2[i], ls2[i]); atomicAdd(&cnt[i], lc[i]); }
  }
}

__global__ void ln_apply_kernel(const float* __restrict__ xcat, const int* __restrict__ batch,
                                const float* __restrict__ s1, const float* __restrict__ s2,
                                const float* __restrict__ cnt, const float* __restrict__ w,
                                const float* __restrict__ b, float* __restrict__ hout, int N) {
  long i = (long)blockIdx.x * blockDim.x + threadIdx.x;
  if (i >= (long)N * DF) return;
  int n = (int)(i / DF), f = (int)(i % DF);
  int g = batch[n];
  float c = cnt[g];
  float norm = fmaxf(c, 1.f) * (float)DF;
  float mean = s1[g] / norm;
  float var  = s2[g] / norm - mean * mean;
  float rs = rsqrtf(var + 1e-5f);
  float y = (xcat[i] - mean) * rs * w[f] + b[f];
  hout[i] = eluf(y);                         // fused elu after layernorm
}

// ---------------- SAG pooling ----------------

__global__ void sag_agg_kernel(const int* __restrict__ sidx, const int* __restrict__ didx,
                               const float* __restrict__ x, float* __restrict__ agg) {
  int f = threadIdx.x;
  if (f >= DF) return;
  long e = blockIdx.x;
  int s = sidx[e], d = didx[e];
  atomicAdd(agg + (long)d * DF + f, x[(long)s * DF + f]);
}

__global__ void sag_score_kernel(const float* __restrict__ agg, const float* __restrict__ x,
                                 const float* __restrict__ Wrel, const float* __restrict__ brel,
                                 const float* __restrict__ Wroot, float* __restrict__ score, int N) {
  int n = blockIdx.x * blockDim.x + threadIdx.x;
  if (n >= N) return;
  const float* ar = agg + (long)n * DF;
  const float* xr = x + (long)n * DF;
  float s = brel[0];
  for (int f = 0; f < DF; ++f) s += ar[f] * Wrel[f] + xr[f] * Wroot[f];
  score[n] = s;
}

__global__ void node_max_kernel(const float* __restrict__ score, const int* __restrict__ batch,
                                float* __restrict__ m, int N) {
  int n = blockIdx.x * blockDim.x + threadIdx.x;
  if (n >= N) return;
  atomicMaxF(m + batch[n], score[n]);
}

__global__ void node_expsum_kernel(float* __restrict__ score, const int* __restrict__ batch,
                                   const float* __restrict__ m, float* __restrict__ z, int N) {
  int n = blockIdx.x * blockDim.x + threadIdx.x;
  if (n >= N) return;
  int g = batch[n];
  float e = __expf(score[n] - m[g]);
  score[n] = e;
  atomicAdd(z + g, e);
}

__global__ void sag_apply_kernel(const float* __restrict__ h, const int* __restrict__ batch,
                                 const float* __restrict__ escore, const float* __restrict__ z,
                                 float* __restrict__ out_nodes, float* __restrict__ out_graph, int N) {
  long i = (long)blockIdx.x * blockDim.x + threadIdx.x;
  if (i >= (long)N * DF) return;
  int n = (int)(i / DF), f = (int)(i % DF);
  int g = batch[n];
  float sc = escore[n] / (z[g] + 1e-16f);
  float v = h[i] * sc;
  out_nodes[i] = v;
  atomicAdd(out_graph + (long)g * DF + f, v);
}

// ---------------- launcher ----------------

extern "C" void kernel_launch(void* const* d_in, const int* in_sizes, int n_in,
                              void* d_out, int out_size, void* d_ws, size_t ws_size,
                              hipStream_t stream) {
  const float* atom_x    = (const float*)d_in[0];
  const int*   atom_ei   = (const int*)d_in[1];
  const int*   atom_batch= (const int*)d_in[2];
  const float* aa_x      = (const float*)d_in[3];
  const int*   aa_ei     = (const int*)d_in[4];
  const int*   aa_batch  = (const int*)d_in[6];
  const int*   m2p_ei    = (const int*)d_in[7];
  const float* Wd = (const float*)d_in[8];
  const float* ad_src = (const float*)d_in[9], *ad_dst = (const float*)d_in[10], *bd = (const float*)d_in[11];
  const float* Wp = (const float*)d_in[12];
  const float* ap_src = (const float*)d_in[13], *ap_dst = (const float*)d_in[14], *bp = (const float*)d_in[15];
  const float* Wi_src = (const float*)d_in[16], *Wi_dst = (const float*)d_in[17];
  const float* ai_src = (const float*)d_in[18], *ai_dst = (const float*)d_in[19], *bi = (const float*)d_in[20];
  const float* ln_d_w = (const float*)d_in[21], *ln_d_b = (const float*)d_in[22];
  const float* ln_p_w = (const float*)d_in[23], *ln_p_b = (const float*)d_in[24];
  const float* pd_Wrel = (const float*)d_in[25], *pd_brel = (const float*)d_in[26], *pd_Wroot = (const float*)d_in[27];
  const float* pp_Wrel = (const float*)d_in[28], *pp_brel = (const float*)d_in[29], *pp_Wroot = (const float*)d_in[30];

  const int N_ATOM = in_sizes[0] / DF;
  const int N_AA   = in_sizes[3] / DF;
  const int E_ATOM = in_sizes[1] / 2;
  const int E_AA   = in_sizes[4] / 2;
  const int E_M2P  = in_sizes[7] / 2;
  const int n_loop = (N_ATOM < N_AA) ? N_ATOM : N_AA;

  char* ws = (char*)d_ws;
  size_t off = 0;
  auto alloc = [&](size_t bytes) -> void* {
    off = (off + 255) & ~(size_t)255;
    void* p = ws + off;
    off += bytes;
    return p;
  };
  auto nb = [](long n, long per) -> unsigned { return (unsigned)((n + per - 1) / per); };

  // scratch (~850 MB total)
  __bf16* atom_exb = (__bf16*)alloc((size_t)N_ATOM * KP * 2);
  __bf16* aa_exb   = (__bf16*)alloc((size_t)N_AA * KP * 2);
  __bf16* atomh_b  = (__bf16*)alloc((size_t)N_ATOM * KP * 2);
  __bf16* wt_d  = (__bf16*)alloc((size_t)NPAD * KP * 2);
  __bf16* wt_p  = (__bf16*)alloc((size_t)NPAD * KP * 2);
  __bf16* wt_is = (__bf16*)alloc((size_t)NPAD * KP * 2);
  __bf16* wt_id = (__bf16*)alloc((size_t)NPAD * KP * 2);
  float* wv_d_s = (float*)alloc(DF * 2 * 4);
  float* wv_d_d = (float*)alloc(DF * 2 * 4);
  float* wv_p_s = (float*)alloc(DF * 2 * 4);
  float* wv_p_d = (float*)alloc(DF * 2 * 4);
  float* wv_i_s = (float*)alloc(DF * 2 * 4);
  float* wv_i_d = (float*)alloc(DF * 2 * 4);
  float* hs_atom_d = (float*)alloc((size_t)N_ATOM * HCF * 4);
  float* hs_aa_i   = (float*)alloc((size_t)N_AA * HCF * 4);
  float* hs_aa_p   = (float*)alloc((size_t)N_AA * HCF * 4);
  float* hs_ath_i  = (float*)alloc((size_t)N_ATOM * HCF * 4);
  float* als_d   = (float*)alloc((size_t)N_ATOM * 2 * 4);
  float* ald_d   = (float*)alloc((size_t)N_ATOM * 2 * 4);
  float* als_ai  = (float*)alloc((size_t)N_AA * 2 * 4);
  float* ald_ai  = (float*)alloc((size_t)N_ATOM * 2 * 4);
  float* als_p   = (float*)alloc((size_t)N_AA * 2 * 4);
  float* ald_p   = (float*)alloc((size_t)N_AA * 2 * 4);
  float* als_ai2 = (float*)alloc((size_t)N_ATOM * 2 * 4);
  float* ald_ai2 = (float*)alloc((size_t)N_AA * 2 * 4);
  float* xcat_atom = (float*)alloc((size_t)N_ATOM * DF * 4);
  float* xcat_aa   = (float*)alloc((size_t)N_AA * DF * 4);
  float* atom_h = (float*)alloc((size_t)N_ATOM * DF * 4);
  float* aa_h   = (float*)alloc((size_t)N_AA * DF * 4);
  float* mbuf = (float*)alloc((size_t)N_AA * 2 * 4);
  float* zbuf = (float*)alloc((size_t)N_AA * 2 * 4);
  float* ebuf = (float*)alloc(((size_t)E_AA + N_AA) * 2 * 4);
  float* s1b  = (float*)alloc(NGR * 4);
  float* s2b  = (float*)alloc(NGR * 4);
  float* cntb = (float*)alloc(NGR * 4);
  float* agg   = (float*)alloc((size_t)N_AA * DF * 4);
  float* score = (float*)alloc((size_t)N_AA * 4);
  float* sm_m  = (float*)alloc(NGR * 4);
  float* sm_z  = (float*)alloc(NGR * 4);

  const float NEG_INF = -__builtin_huge_valf();

  // 1. elu + bf16 cast (padded)
  cast_pad_kernel<<<nb((long)N_ATOM * KP, 256), 256, 0, stream>>>(atom_x, atom_exb, N_ATOM, 1);
  cast_pad_kernel<<<nb((long)N_AA * KP, 256), 256, 0, stream>>>(aa_x, aa_exb, N_AA, 1);

  // 2. weight prep: transposed bf16 tiles + fused attention vectors
  prep_wt_kernel<<<nb(NPAD * KP, 256), 256, 0, stream>>>(Wd, wt_d);
  prep_wt_kernel<<<nb(NPAD * KP, 256), 256, 0, stream>>>(Wp, wt_p);
  prep_wt_kernel<<<nb(NPAD * KP, 256), 256, 0, stream>>>(Wi_src, wt_is);
  prep_wt_kernel<<<nb(NPAD * KP, 256), 256, 0, stream>>>(Wi_dst, wt_id);
  prep_wv_kernel<<<nb(DF * 2, 256), 256, 0, stream>>>(Wd, ad_src, wv_d_s);
  prep_wv_kernel<<<nb(DF * 2, 256), 256, 0, stream>>>(Wd, ad_dst, wv_d_d);
  prep_wv_kernel<<<nb(DF * 2, 256), 256, 0, stream>>>(Wp, ap_src, wv_p_s);
  prep_wv_kernel<<<nb(DF * 2, 256), 256, 0, stream>>>(Wp, ap_dst, wv_p_d);
  prep_wv_kernel<<<nb(DF * 2, 256), 256, 0, stream>>>(Wi_src, ai_src, wv_i_s);
  prep_wv_kernel<<<nb(DF * 2, 256), 256, 0, stream>>>(Wi_dst, ai_dst, wv_i_d);

  // 3. WMMA GEMMs (hs projections only; hd folded into al matvecs)
  gemm_bf16_kernel<<<nb(N_ATOM, 64), 128, 0, stream>>>(atom_exb, wt_d, hs_atom_d, N_ATOM);
  gemm_bf16_kernel<<<nb(N_AA, 64), 128, 0, stream>>>(aa_exb, wt_is, hs_aa_i, N_AA);
  gemm_bf16_kernel<<<nb(N_AA, 64), 128, 0, stream>>>(aa_exb, wt_p, hs_aa_p, N_AA);

  // 4. attention logit vectors (matvec with elu fused on raw inputs)
  al_kernel<<<nb(2L * N_ATOM, 256), 256, 0, stream>>>(atom_x, wv_d_s, als_d, N_ATOM, 1);
  al_kernel<<<nb(2L * N_ATOM, 256), 256, 0, stream>>>(atom_x, wv_d_d, ald_d, N_ATOM, 1);
  al_kernel<<<nb(2L * N_AA, 256), 256, 0, stream>>>(aa_x, wv_i_s, als_ai, N_AA, 1);
  al_kernel<<<nb(2L * N_ATOM, 256), 256, 0, stream>>>(atom_x, wv_i_d, ald_ai, N_ATOM, 1);
  al_kernel<<<nb(2L * N_AA, 256), 256, 0, stream>>>(aa_x, wv_p_s, als_p, N_AA, 1);
  al_kernel<<<nb(2L * N_AA, 256), 256, 0, stream>>>(aa_x, wv_p_d, ald_p, N_AA, 1);
  al_kernel<<<nb(2L * N_AA, 256), 256, 0, stream>>>(aa_x, wv_i_d, ald_ai2, N_AA, 1);

  auto run_gat = [&](const int* sidx, const int* didx, int E, int nl,
                     const float* als, const float* ald, const float* hs,
                     int n_dst, float* outp, int col) {
    long Et = (long)E + nl;
    fill_kernel<<<nb((long)n_dst * 2, 256), 256, 0, stream>>>(mbuf, NEG_INF, (long)n_dst * 2);
    fill_kernel<<<nb((long)n_dst * 2, 256), 256, 0, stream>>>(zbuf, 0.f, (long)n_dst * 2);
    edge_logit_max_kernel<<<nb(Et * 2, 256), 256, 0, stream>>>(sidx, didx, E, nl, als, ald, ebuf, mbuf);
    edge_expsum_kernel<<<nb(Et * 2, 256), 256, 0, stream>>>(didx, E, nl, ebuf, mbuf, zbuf);
    edge_agg_kernel<<<nb(Et, 2), 256, 0, stream>>>(sidx, didx, E, nl, ebuf, zbuf, hs, outp, col);
  };

  // 5. atom-side GATs -> xcat_atom [N_ATOM, 200] (bias pre-filled)
  init_xcat_kernel<<<nb((long)N_ATOM * DF, 256), 256, 0, stream>>>(xcat_atom, bd, bi, (long)N_ATOM * DF);
  run_gat(atom_ei, atom_ei + E_ATOM, E_ATOM, N_ATOM, als_d, ald_d, hs_atom_d, N_ATOM, xcat_atom, 0);
  run_gat(m2p_ei + E_M2P, m2p_ei, E_M2P, n_loop, als_ai, ald_ai, hs_aa_i, N_ATOM, xcat_atom, HCF);

  // 6. graph layernorm (drug) + elu -> atom_h; re-cast for next GEMM
  fill_kernel<<<nb(NGR, 256), 256, 0, stream>>>(s1b, 0.f, NGR);
  fill_kernel<<<nb(NGR, 256), 256, 0, stream>>>(s2b, 0.f, NGR);
  fill_kernel<<<nb(NGR, 256), 256, 0, stream>>>(cntb, 0.f, NGR);
  ln_stats_kernel<<<nb(N_ATOM, 256), 256, 0, stream>>>(xcat_atom, atom_batch, N_ATOM, s1b, s2b, cntb);
  ln_apply_kernel<<<nb((long)N_ATOM * DF, 256), 256, 0, stream>>>(xcat_atom, atom_batch, s1b, s2b, cntb,
                                                                  ln_d_w, ln_d_b, atom_h, N_ATOM);
  cast_pad_kernel<<<nb((long)N_ATOM * KP, 256), 256, 0, stream>>>(atom_h, atomh_b, N_ATOM, 0);

  // 7. updated-atom projection for aa_inter
  gemm_bf16_kernel<<<nb(N_ATOM, 64), 128, 0, stream>>>(atomh_b, wt_is, hs_ath_i, N_ATOM);
  al_kernel<<<nb(2L * N_ATOM, 256), 256, 0, stream>>>(atom_h, wv_i_s, als_ai2, N_ATOM, 0);

  // 8. protein-side GATs -> xcat_aa
  init_xcat_kernel<<<nb((long)N_AA * DF, 256), 256, 0, stream>>>(xcat_aa, bp, bi, (long)N_AA * DF);
  run_gat(aa_ei, aa_ei + E_AA, E_AA, N_AA, als_p, ald_p, hs_aa_p, N_AA, xcat_aa, 0);
  run_gat(m2p_ei, m2p_ei + E_M2P, E_M2P, n_loop, als_ai2, ald_ai2, hs_ath_i, N_AA, xcat_aa, HCF);

  // 9. graph layernorm (protein) + elu -> aa_h
  fill_kernel<<<nb(NGR, 256), 256, 0, stream>>>(s1b, 0.f, NGR);
  fill_kernel<<<nb(NGR, 256), 256, 0, stream>>>(s2b, 0.f, NGR);
  fill_kernel<<<nb(NGR, 256), 256, 0, stream>>>(cntb, 0.f, NGR);
  ln_stats_kernel<<<nb(N_AA, 256), 256, 0, stream>>>(xcat_aa, aa_batch, N_AA, s1b, s2b, cntb);
  ln_apply_kernel<<<nb((long)N_AA * DF, 256), 256, 0, stream>>>(xcat_aa, aa_batch, s1b, s2b, cntb,
                                                                ln_p_w, ln_p_b, aa_h, N_AA);

  // 10. outputs
  float* out = (float*)d_out;
  float* atom_out = out;
  float* aa_out   = out + (size_t)N_ATOM * DF;
  float* drug_g   = aa_out + (size_t)N_AA * DF;
  float* prot_g   = drug_g + (size_t)NGR * DF;
  fill_kernel<<<nb(2L * NGR * DF, 256), 256, 0, stream>>>(drug_g, 0.f, 2L * NGR * DF);

  auto run_sag = [&](const float* h, const int* sidx, const int* didx, int E, const int* batch,
                     const float* Wrel, const float* brel, const float* Wroot, int N,
                     float* out_nodes, float* out_graph) {
    fill_kernel<<<nb((long)N * DF, 256), 256, 0, stream>>>(agg, 0.f, (long)N * DF);
    sag_agg_kernel<<<(unsigned)E, 256, 0, stream>>>(sidx, didx, h, agg);
    sag_score_kernel<<<nb(N, 256), 256, 0, stream>>>(agg, h, Wrel, brel, Wroot, score, N);
    fill_kernel<<<nb(NGR, 256), 256, 0, stream>>>(sm_m, NEG_INF, NGR);
    fill_kernel<<<nb(NGR, 256), 256, 0, stream>>>(sm_z, 0.f, NGR);
    node_max_kernel<<<nb(N, 256), 256, 0, stream>>>(score, batch, sm_m, N);
    node_expsum_kernel<<<nb(N, 256), 256, 0, stream>>>(score, batch, sm_m, sm_z, N);
    sag_apply_kernel<<<nb((long)N * DF, 256), 256, 0, stream>>>(h, batch, score, sm_z,
                                                                out_nodes, out_graph, N);
  };

  run_sag(atom_h, atom_ei, atom_ei + E_ATOM, E_ATOM, atom_batch, pd_Wrel, pd_brel, pd_Wroot,
          N_ATOM, atom_out, drug_g);
  run_sag(aa_h, aa_ei, aa_ei + E_AA, E_AA, aa_batch, pp_Wrel, pp_brel, pp_Wroot,
          N_AA, aa_out, prot_g);
}
